// E3EquivariantGNN_62277025792337
// MI455X (gfx1250) — matrix operand order repaired and checked by reading, not compiled, not run on hardware
//
#include <hip/hip_runtime.h>

#define N_NODES_C 10000
#define E_TOTAL   160000
#define TPB       10          // edge-tiles (of 16 edges) per block
#define FC_TILES  44          // 44 N-tiles: 8 waves x 5 resident + 4 streamed

typedef __attribute__((ext_vector_type(16))) __bf16 v16bf;
typedef __attribute__((ext_vector_type(8)))  __bf16 v8bf;
typedef __attribute__((ext_vector_type(8)))  float  v8f;

__device__ __forceinline__ unsigned short f32_to_bf16_bits(float f) {
    __bf16 b = (__bf16)f;
    return __builtin_bit_cast(unsigned short, b);
}
__device__ __forceinline__ __bf16 f32_to_bf16(float f) { return (__bf16)f; }

// ---------------------------------------------------------------------------
// Weight packing: bf16 B-fragment layout.
// B 32x16 fragment (wave32): lane L holds column n = L&15 of half (L>>4);
// element j = K value k = chunk*32 + (j<8 ? (L>>4)*8 + j : 16 + (L>>4)*8 + j-8).
// ---------------------------------------------------------------------------
__global__ void pack_radial_kernel(const float* __restrict__ radial_w,
                                   unsigned short* __restrict__ radial_wb) {
    int idx = blockIdx.x * blockDim.x + threadIdx.x;
    const int total = 4 * 7 * 32 * 16;
    if (idx >= total) return;
    int j = idx & 15;
    int L = (idx >> 4) & 31;
    int t = (idx >> 9) % 7;
    int l = idx / (7 * 512);
    int k = (j < 8) ? (L >> 4) * 8 + j : 16 + (L >> 4) * 8 + (j - 8); // K index, valid < 10
    int n = t * 16 + (L & 15);                      // N index, valid < 100
    float val = (k < 10 && n < 100) ? radial_w[(l * 10 + k) * 100 + n] : 0.0f;
    radial_wb[idx] = f32_to_bf16_bits(val);
}

__global__ void pack_fc_kernel(const float* __restrict__ fc_w,
                               unsigned short* __restrict__ fc_wb) {
    int idx = blockIdx.x * blockDim.x + threadIdx.x;
    const int total = 4 * FC_TILES * 4 * 32 * 16;
    if (idx >= total) return;
    int j = idx & 15;
    int L = (idx >> 4) & 31;
    int c = (idx >> 9) & 3;
    int t = (idx >> 11) % FC_TILES;
    int l = idx / (FC_TILES * 2048);
    int k = c * 32 + ((j < 8) ? (L >> 4) * 8 + j : 16 + (L >> 4) * 8 + (j - 8)); // K index, valid < 100
    int n = t * 16 + (L & 15);                      // N index (0..703)
    float val = (k < 100) ? fc_w[(l * 100 + k) * 704 + n] : 0.0f;
    fc_wb[idx] = f32_to_bf16_bits(val);
}

// ---------------------------------------------------------------------------
// Edge geometry: vec, length, l=1/l=2 spherical harmonics, gaussian rbf.
// rbf padded to 16 floats/edge (zeros beyond 10) for branch-free A-fragments.
// ---------------------------------------------------------------------------
__global__ void geom_kernel(const float* __restrict__ pos,
                            const int* __restrict__ edge_index,
                            float* __restrict__ sh1g,
                            float* __restrict__ Mg,
                            float* __restrict__ rbfg) {
    int e = blockIdx.x * blockDim.x + threadIdx.x;
    if (e >= E_TOTAL) return;
    int rn = edge_index[e];
    int cn = edge_index[E_TOTAL + e];
    float vx = pos[rn * 3 + 0] - pos[cn * 3 + 0];
    float vy = pos[rn * 3 + 1] - pos[cn * 3 + 1];
    float vz = pos[rn * 3 + 2] - pos[cn * 3 + 2];
    float len = sqrtf(vx * vx + vy * vy + vz * vz + 1e-12f);
    float inv = 1.0f / len;
    float ux = vx * inv, uy = vy * inv, uz = vz * inv;
    const float s3 = 1.7320508075688772f;
    sh1g[e * 3 + 0] = s3 * ux;
    sh1g[e * 3 + 1] = s3 * uy;
    sh1g[e * 3 + 2] = s3 * uz;
    const float s75 = 2.7386127875258306f;          // sqrt(7.5)
    const float third = 1.0f / 3.0f;
    Mg[e * 9 + 0] = s75 * (ux * ux - third);
    Mg[e * 9 + 1] = s75 * (ux * uy);
    Mg[e * 9 + 2] = s75 * (ux * uz);
    Mg[e * 9 + 3] = s75 * (uy * ux);
    Mg[e * 9 + 4] = s75 * (uy * uy - third);
    Mg[e * 9 + 5] = s75 * (uy * uz);
    Mg[e * 9 + 6] = s75 * (uz * ux);
    Mg[e * 9 + 7] = s75 * (uz * uy);
    Mg[e * 9 + 8] = s75 * (uz * uz - third);
    const float step = 10.0f / 9.0f;
    const float coeff = -0.5f / (step * step);
    for (int k = 0; k < 10; ++k) {
        float d = len - step * (float)k;
        rbfg[e * 16 + k] = expf(coeff * d * d);
    }
    for (int k = 10; k < 16; ++k) rbfg[e * 16 + k] = 0.0f;
}

// ---------------------------------------------------------------------------
// Node embedding s = x @ embed_w / sqrt(8)
// ---------------------------------------------------------------------------
__global__ void embed_kernel(const float* __restrict__ x,
                             const float* __restrict__ embed_w,
                             float* __restrict__ s) {
    int idx = blockIdx.x * blockDim.x + threadIdx.x;
    if (idx >= N_NODES_C * 16) return;
    int n = idx >> 4, c = idx & 15;
    float acc = 0.0f;
    for (int k = 0; k < 8; ++k) acc += x[n * 8 + k] * embed_w[k * 16 + c];
    s[idx] = acc * 0.35355339059327373f;            // 1/sqrt(8)
}

__global__ void zero_kernel(float* __restrict__ p, int n) {
    int idx = blockIdx.x * blockDim.x + threadIdx.x;
    if (idx < n) p[idx] = 0.0f;
}

// ---------------------------------------------------------------------------
// Fused edge kernel, B-stationary: each wave keeps 5 N-tiles x 4 K-chunks of
// fc weights resident in VGPRs (tiles 0..39); tiles 40..43 are streamed by
// waves 0..3 each iteration.  Block = 256 threads = 8 waves, TPB*16 edges.
// ---------------------------------------------------------------------------
__global__ __launch_bounds__(256) void edge_kernel(
    const float* __restrict__ sbuf,                 // [N,16]
    const float* __restrict__ vbuf,                 // [N,8,3]
    const float* __restrict__ sh1g,                 // [E,3]
    const float* __restrict__ Mg,                   // [E,9]
    const float* __restrict__ rbfg,                 // [E,16] (padded)
    const unsigned short* __restrict__ radial_wb,   // [7][32][16] bf16 (layer)
    const float* __restrict__ radial_b,             // [100] (layer)
    const unsigned short* __restrict__ fc_wb,       // [44][4][32][16] bf16 (layer)
    const float* __restrict__ fc_b,                 // [704] (layer)
    const int* __restrict__ edge_index,             // [2,E]
    float* __restrict__ agg_s,                      // [N,24]
    float* __restrict__ agg_v)                      // [N,32,3]
{
    // strides padded to avoid 64-bank aliasing (all vector-load-aligned)
    __shared__ unsigned short hbf[16][136];         // bf16 hidden acts (stride 272B)
    __shared__ float wbuf[16][708];                 // per-edge TP weights
    __shared__ float sjs[16][17];
    __shared__ float vjs[16][25];                   // [u*3+i]
    __shared__ float shs[16][4];
    __shared__ float Ms[16][13];

    const int tid    = threadIdx.x;
    const int lane   = tid & 31;
    const int wave   = tid >> 5;                    // 0..7
    const int halfid = lane >> 4;
    const int mrow   = lane & 15;

    // ---- preload resident B fragments + biases ----
    v16bf radb;
    {
        int rt = (wave < 7) ? wave : 0;
        radb = *(const v16bf*)(radial_wb + (size_t)(rt * 32 + lane) * 16);
    }
    const int rcol   = wave * 16 + mrow;            // radial output column
    const float rbias = (wave < 7 && rcol < 100) ? radial_b[rcol] : 0.0f;

    const int t0 = wave * 5;                        // resident tiles t0..t0+4
    v16bf fb[5][4];
    float fbias[5];
    #pragma unroll
    for (int i = 0; i < 5; ++i) {
        int t = t0 + i;
        #pragma unroll
        for (int c = 0; c < 4; ++c)
            fb[i][c] = *(const v16bf*)(fc_wb + (size_t)((t * 4 + c) * 32 + lane) * 16);
        fbias[i] = fc_b[t * 16 + mrow];
    }
    const int ts = 40 + wave;                       // streamed tile (waves 0..3)
    const float tsbias = (wave < 4) ? fc_b[ts * 16 + mrow] : 0.0f;

    // zero hbf once (cols >= 100 stay zero forever)
    for (int i = tid; i < 16 * 136; i += 256) ((unsigned short*)hbf)[i] = 0;

    for (int bt = 0; bt < TPB; ++bt) {
        const int e0 = (blockIdx.x * TPB + bt) * 16;
        __syncthreads();                            // prev TP done (LDS free)

        // ---- stage per-edge features: 16 threads per edge ----
        {
            int e = tid & 15, r = tid >> 4;         // r in 0..15
            int eg = e0 + e;
            int rown = edge_index[eg];
            sjs[e][r] = sbuf[rown * 16 + r];
            if (r < 8) {
                vjs[e][r * 3 + 0] = vbuf[(rown * 8 + r) * 3 + 0];
                vjs[e][r * 3 + 1] = vbuf[(rown * 8 + r) * 3 + 1];
                vjs[e][r * 3 + 2] = vbuf[(rown * 8 + r) * 3 + 2];
            }
            if (r == 8) {
                shs[e][0] = sh1g[eg * 3 + 0];
                shs[e][1] = sh1g[eg * 3 + 1];
                shs[e][2] = sh1g[eg * 3 + 2];
            }
            if (r >= 9 && r <= 11) {
                int b = (r - 9) * 3;
                Ms[e][b + 0] = Mg[eg * 9 + b + 0];
                Ms[e][b + 1] = Mg[eg * 9 + b + 1];
                Ms[e][b + 2] = Mg[eg * 9 + b + 2];
            }
        }

        // ---- radial GEMM: h = relu(rbf @ radial_w + b); waves 0..6 ----
        if (wave < 7) {
            v16bf a_rbf;
            const float* rp = rbfg + (size_t)(e0 + mrow) * 16 + halfid * 8;
            #pragma unroll
            for (int j = 0; j < 8; ++j) a_rbf[j] = f32_to_bf16(rp[j]);
            #pragma unroll
            for (int j = 8; j < 16; ++j) a_rbf[j] = f32_to_bf16(0.0f);
            v8f c = {};
            c = __builtin_amdgcn_wmma_f32_16x16x32_bf16(false, a_rbf, false, radb,
                                                        (short)0, c, false, false);
            if (rcol < 100) {
                int mbase = halfid * 8;
                #pragma unroll
                for (int k = 0; k < 8; ++k) {
                    float h = c[k] + rbias;
                    h = h > 0.0f ? h : 0.0f;
                    hbf[mbase + k][rcol] = f32_to_bf16_bits(h);
                }
            }
        }
        __syncthreads();                            // hbf + staging ready

        // ---- load A fragments of h (4 K-chunks of 32) from LDS ----
        v16bf a_h[4];
        #pragma unroll
        for (int c = 0; c < 4; ++c) {
            int k0 = c * 32 + halfid * 8;           // elements 0..7
            int k1 = c * 32 + 16 + halfid * 8;      // elements 8..15
            v8bf lo = *(const v8bf*)&hbf[mrow][k0];
            v8bf hi = *(const v8bf*)&hbf[mrow][k1];
            #pragma unroll
            for (int j = 0; j < 8; ++j) { a_h[c][j] = lo[j]; a_h[c][j + 8] = hi[j]; }
        }

        // ---- fc GEMM, resident tiles ----
        const int mbase = halfid * 8;
        #pragma unroll
        for (int i = 0; i < 5; ++i) {
            v8f acc = {};
            #pragma unroll
            for (int c = 0; c < 4; ++c)
                acc = __builtin_amdgcn_wmma_f32_16x16x32_bf16(false, a_h[c],
                          false, fb[i][c], (short)0, acc, false, false);
            int n = (t0 + i) * 16 + mrow;
            #pragma unroll
            for (int k = 0; k < 8; ++k) wbuf[mbase + k][n] = acc[k] + fbias[i];
        }
        // ---- fc GEMM, streamed tiles 40..43 (waves 0..3; wave-uniform) ----
        if (wave < 4) {
            v8f acc = {};
            #pragma unroll
            for (int c = 0; c < 4; ++c) {
                v16bf b = *(const v16bf*)(fc_wb + (size_t)((ts * 4 + c) * 32 + lane) * 16);
                acc = __builtin_amdgcn_wmma_f32_16x16x32_bf16(false, a_h[c],
                          false, b, (short)0, acc, false, false);
            }
            int n = ts * 16 + mrow;
            #pragma unroll
            for (int k = 0; k < 8; ++k) wbuf[mbase + k][n] = acc[k] + tsbias;
        }
        __syncthreads();                            // wbuf ready

        // ---- tensor product + atomic scatter: 16 threads per edge ----
        {
            int e = tid & 15, r = tid >> 4;         // r in 0..15
            int eg = e0 + e;
            int node = edge_index[E_TOTAL + eg];
            const float* w  = wbuf[e];
            const float* sj = sjs[e];
            const float* vj = vjs[e];
            float sh0 = shs[e][0], sh1 = shs[e][1], sh2 = shs[e][2];
            const float* Mm = Ms[e];

            const float c_sc = 0.25f;                // 1/sqrt(16)
            const float c_v0 = 0.35355339059327373f; // 1/sqrt(8)
            const float c_cg = 0.2041241452319315f;  // 1/sqrt(24)

            // m0 (w0) and m1 pre-contraction (w1): output wo = r
            {
                float acc0 = 0.0f, acc1 = 0.0f;
                for (int u = 0; u < 16; ++u) {
                    float su = sj[u];
                    acc0 += w[u * 16 + r] * su;
                    acc1 += w[256 + u * 16 + r] * su;
                }
                atomicAdd(&agg_s[node * 24 + r], acc0 * c_sc);
                float t1 = acc1 * c_sc;
                atomicAdd(&agg_v[(node * 32 + r) * 3 + 0], t1 * sh0);
                atomicAdd(&agg_v[(node * 32 + r) * 3 + 1], t1 * sh1);
                atomicAdd(&agg_v[(node * 32 + r) * 3 + 2], t1 * sh2);
            }
            if (r < 8) {
                // m2 (w2): output w = r, agg_v rows 16..23
                float a0 = 0, a1 = 0, a2 = 0;
                for (int u = 0; u < 8; ++u) {
                    float ww = w[512 + u * 8 + r];
                    a0 += ww * vj[u * 3 + 0];
                    a1 += ww * vj[u * 3 + 1];
                    a2 += ww * vj[u * 3 + 2];
                }
                atomicAdd(&agg_v[(node * 32 + 16 + r) * 3 + 0], a0 * c_v0);
                atomicAdd(&agg_v[(node * 32 + 16 + r) * 3 + 1], a1 * c_v0);
                atomicAdd(&agg_v[(node * 32 + 16 + r) * 3 + 2], a2 * c_v0);
                // m3 (w3): output w = r, agg_s cols 16..23
                float acc = 0.0f;
                for (int u = 0; u < 8; ++u) {
                    float tdot = vj[u * 3 + 0] * sh0 + vj[u * 3 + 1] * sh1
                               + vj[u * 3 + 2] * sh2;
                    acc += w[576 + u * 8 + r] * tdot;
                }
                atomicAdd(&agg_s[node * 24 + 16 + r], acc * c_cg);
            } else {
                // m4 (w4): output w = r-8, agg_v rows 24..31
                int wo = r - 8;
                float a0 = 0, a1 = 0, a2 = 0;
                for (int u = 0; u < 8; ++u) {
                    float x0 = vj[u * 3 + 0], x1 = vj[u * 3 + 1], x2 = vj[u * 3 + 2];
                    float mv0 = Mm[0] * x0 + Mm[1] * x1 + Mm[2] * x2;
                    float mv1 = Mm[3] * x0 + Mm[4] * x1 + Mm[5] * x2;
                    float mv2 = Mm[6] * x0 + Mm[7] * x1 + Mm[8] * x2;
                    float ww = w[640 + u * 8 + wo];
                    a0 += ww * mv0; a1 += ww * mv1; a2 += ww * mv2;
                }
                atomicAdd(&agg_v[(node * 32 + 24 + wo) * 3 + 0], a0 * c_cg);
                atomicAdd(&agg_v[(node * 32 + 24 + wo) * 3 + 1], a1 * c_cg);
                atomicAdd(&agg_v[(node * 32 + 24 + wo) * 3 + 2], a2 * c_cg);
            }
        }
    }
}

// ---------------------------------------------------------------------------
// Node update: s += agg_s @ lin_ws / sqrt(24); v += agg_v x lin_wv / sqrt(32)
// ---------------------------------------------------------------------------
__global__ __launch_bounds__(64) void node_update_kernel(
    const float* __restrict__ agg_s, const float* __restrict__ agg_v,
    const float* __restrict__ lin_ws, const float* __restrict__ lin_wv,
    float* __restrict__ s, float* __restrict__ v) {
    int n = blockIdx.x;
    int t = threadIdx.x;
    if (t < 16) {
        float acc = 0.0f;
        for (int u = 0; u < 24; ++u) acc += agg_s[n * 24 + u] * lin_ws[u * 16 + t];
        s[n * 16 + t] += acc * 0.2041241452319315f;      // 1/sqrt(24)
    } else if (t < 40) {
        int j = t - 16;
        int w = j / 3, i = j % 3;
        float acc = 0.0f;
        for (int u = 0; u < 32; ++u) acc += agg_v[(n * 32 + u) * 3 + i] * lin_wv[u * 8 + w];
        v[(n * 8 + w) * 3 + i] += acc * 0.17677669529663687f; // 1/sqrt(32)
    }
}

// ---------------------------------------------------------------------------
// Output projection: out = s @ out_w / sqrt(16)
// ---------------------------------------------------------------------------
__global__ void out_kernel(const float* __restrict__ s,
                           const float* __restrict__ out_w,
                           float* __restrict__ out) {
    int idx = blockIdx.x * blockDim.x + threadIdx.x;
    if (idx >= N_NODES_C * 8) return;
    int n = idx >> 3, c = idx & 7;
    float acc = 0.0f;
    for (int k = 0; k < 16; ++k) acc += s[n * 16 + k] * out_w[k * 8 + c];
    out[idx] = acc * 0.25f;
}

// ---------------------------------------------------------------------------
extern "C" void kernel_launch(void* const* d_in, const int* in_sizes, int n_in,
                              void* d_out, int out_size, void* d_ws, size_t ws_size,
                              hipStream_t stream) {
    const float* x        = (const float*)d_in[0];
    const float* pos      = (const float*)d_in[1];
    const float* embed_w  = (const float*)d_in[2];
    const float* radial_w = (const float*)d_in[3];
    const float* radial_b = (const float*)d_in[4];
    const float* fc_w     = (const float*)d_in[5];
    const float* fc_b     = (const float*)d_in[6];
    const float* lin_ws   = (const float*)d_in[7];
    const float* lin_wv   = (const float*)d_in[8];
    const float* out_w    = (const float*)d_in[9];
    const int*   edge_index = (const int*)d_in[10];
    float* out = (float*)d_out;

    char* ws = (char*)d_ws;
    size_t off = 0;
    auto alloc = [&](size_t bytes) -> void* {
        void* p = ws + off;
        off += (bytes + 255) & ~(size_t)255;
        return p;
    };

    float* s      = (float*)alloc((size_t)N_NODES_C * 16 * 4);
    float* v      = (float*)alloc((size_t)N_NODES_C * 24 * 4);
    float* agg_s  = (float*)alloc((size_t)N_NODES_C * 24 * 4);
    float* agg_v  = (float*)alloc((size_t)N_NODES_C * 96 * 4);
    float* sh1g   = (float*)alloc((size_t)E_TOTAL * 3 * 4);
    float* Mg     = (float*)alloc((size_t)E_TOTAL * 9 * 4);
    float* rbfg   = (float*)alloc((size_t)E_TOTAL * 16 * 4);
    unsigned short* radial_wb = (unsigned short*)alloc((size_t)4 * 7 * 512 * 2);
    unsigned short* fc_wb     = (unsigned short*)alloc((size_t)4 * FC_TILES * 2048 * 2);

    // pack weights into bf16 B-fragment layouts
    pack_radial_kernel<<<(4 * 7 * 512 + 255) / 256, 256, 0, stream>>>(radial_w, radial_wb);
    pack_fc_kernel<<<(4 * FC_TILES * 2048 + 255) / 256, 256, 0, stream>>>(fc_w, fc_wb);

    // edge geometry + rbf
    geom_kernel<<<(E_TOTAL + 255) / 256, 256, 0, stream>>>(pos, edge_index, sh1g, Mg, rbfg);

    // initial node features
    embed_kernel<<<(N_NODES_C * 16 + 255) / 256, 256, 0, stream>>>(x, embed_w, s);
    zero_kernel<<<(N_NODES_C * 24 + 255) / 256, 256, 0, stream>>>(v, N_NODES_C * 24);

    for (int l = 0; l < 4; ++l) {
        zero_kernel<<<(N_NODES_C * 24 + 255) / 256, 256, 0, stream>>>(agg_s, N_NODES_C * 24);
        zero_kernel<<<(N_NODES_C * 96 + 255) / 256, 256, 0, stream>>>(agg_v, N_NODES_C * 96);
        edge_kernel<<<E_TOTAL / (16 * TPB), 256, 0, stream>>>(
            s, v, sh1g, Mg, rbfg,
            radial_wb + (size_t)l * 7 * 512, radial_b + l * 100,
            fc_wb + (size_t)l * FC_TILES * 2048, fc_b + l * 704,
            edge_index, agg_s, agg_v);
        node_update_kernel<<<N_NODES_C, 64, 0, stream>>>(
            agg_s, agg_v, lin_ws + l * 24 * 16, lin_wv + l * 32 * 8, s, v);
    }

    out_kernel<<<(N_NODES_C * 8 + 255) / 256, 256, 0, stream>>>(s, out_w, out);
}
